// ANIModel_4698694222407
// MI455X (gfx1250) — matrix operand order repaired and checked by reading, compiled, not run on hardware
//
#include <hip/hip_runtime.h>
#include <hip/hip_bf16.h>
#include <stdint.h>

// ---------------------------------------------------------------------------
// ANI per-type MLP (MoE, 4 experts) on MI455X via bf16 WMMA (f32 accumulate)
// ---------------------------------------------------------------------------

typedef __bf16 v16bf __attribute__((ext_vector_type(16)));
typedef float  v8f   __attribute__((ext_vector_type(8)));
typedef unsigned short us8 __attribute__((ext_vector_type(8)));

#define NTYPES 4
#define AEVD   384
#define H0 160
#define H1 128
#define H2 96
#define CELU_ALPHA 0.1f

// Packed bf16 weight blob offsets (ushort units) within one type's blob.
// Layout per (layer): [kt][nt][lane(32)][16 halves], 512 halves per 32x16 tile.
#define PK_L0 0           // 12 kt * 10 nt * 512
#define PK_L1 61440       // + 5 kt *  8 nt * 512
#define PK_L2 81920       // + 4 kt *  6 nt * 512
#define PK_L3 94208       // + 3 kt *  1 nt * 512
#define PK_T  95744

__device__ __forceinline__ unsigned short f2bf(float f) {
  unsigned int u = __float_as_uint(f);
  unsigned int r = 0x7FFFu + ((u >> 16) & 1u);   // round-to-nearest-even
  return (unsigned short)((u + r) >> 16);
}

__device__ __forceinline__ float celu01(float x) {
  return x > 0.f ? x : CELU_ALPHA * (__expf(x * (1.f / CELU_ALPHA)) - 1.f);
}

union BF16Frag { v16bf v; us8 h[2]; };

// ---------------------------------------------------------------------------
// Kernel 0: zero the per-type counters
// ---------------------------------------------------------------------------
__global__ void init_counts_kernel(int* counts) {
  if (threadIdx.x < NTYPES) counts[threadIdx.x] = 0;
}

// ---------------------------------------------------------------------------
// Kernel 1: bucket atoms by species.
// Per-block LDS histogram + ONE global atomicAdd per (block, type):
// 131072 contended global RMWs -> <= 2048, rest are pipelined DS atomics.
// ---------------------------------------------------------------------------
__global__ void bucket_kernel(const long long* __restrict__ species,
                              int* __restrict__ counts,
                              int* __restrict__ lists, int natoms) {
  __shared__ int lcnt[NTYPES];
  __shared__ int lbase[NTYPES];
  const int tid = threadIdx.x;
  if (tid < NTYPES) lcnt[tid] = 0;
  __syncthreads();

  const int i = blockIdx.x * blockDim.x + tid;
  int t = 0, rank = 0;
  bool valid = (i < natoms);
  if (valid) {
    t = (int)species[i];
    rank = atomicAdd(&lcnt[t], 1);        // LDS atomic: block-local rank
  }
  __syncthreads();

  if (tid < NTYPES) lbase[tid] = atomicAdd(&counts[tid], lcnt[tid]);
  __syncthreads();

  if (valid) lists[t * natoms + lbase[t] + rank] = i;
}

// ---------------------------------------------------------------------------
// Kernel 2: pack f32 weights into bf16 B-fragment layout
//   packed[t][layer][kt][nt][lane][h] = W_l[t][kt*32 + 16*(h>=8) + 8*(lane>=16) + (h&7)]
//                                           [nt*16 + (lane&15)]
// ---------------------------------------------------------------------------
__global__ void pack_weights_kernel(const float* __restrict__ W0,
                                    const float* __restrict__ W1,
                                    const float* __restrict__ W2,
                                    const float* __restrict__ W3,
                                    unsigned short* __restrict__ wpk) {
  int id = blockIdx.x * blockDim.x + threadIdx.x;
  if (id >= NTYPES * PK_T) return;
  int t = id / PK_T;
  int rem = id - t * PK_T;
  int Kd, realN, ntiles, off;
  const float* Ws;
  if (rem < PK_L1)      { Kd = AEVD; realN = H0; ntiles = H0 / 16; off = PK_L0; Ws = W0; }
  else if (rem < PK_L2) { Kd = H0;   realN = H1; ntiles = H1 / 16; off = PK_L1; Ws = W1; }
  else if (rem < PK_L3) { Kd = H1;   realN = H2; ntiles = H2 / 16; off = PK_L2; Ws = W2; }
  else                  { Kd = H2;   realN = 1;  ntiles = 1;       off = PK_L3; Ws = W3; }
  int u    = rem - off;
  int h    = u & 15;
  int lane = (u >> 4) & 31;
  int tile = u >> 9;
  int nt   = tile % ntiles;
  int kt   = tile / ntiles;
  int K = kt * 32 + (h >> 3) * 16 + ((lane >> 4) & 1) * 8 + (h & 7);
  int N = nt * 16 + (lane & 15);
  float v = 0.f;
  if (N < realN) v = Ws[(size_t)t * Kd * realN + (size_t)K * realN + N];
  wpk[id] = f2bf(v);
}

// ---------------------------------------------------------------------------
// One MLP layer: [16 x KD] (LDS bf16) @ [KD x ND] (packed bf16) + bias, CELU,
// write [16 x ND] bf16 back to LDS. One wave32, WMMA f32_16x16x32_bf16.
// nt unrolled x2 -> two independent accumulator chains for XDL pipelining.
// ---------------------------------------------------------------------------
template <int KD, int ND>
__device__ __forceinline__ void mlp_layer(const unsigned short* xin,
                                          unsigned short* yout,
                                          const unsigned short* __restrict__ wp,
                                          const float* __restrict__ bias,
                                          int lane) {
  const int nlow  = lane & 15;
  const int lhalf = (lane >> 4) & 1;
  // A-fragment: lane holds row M=lane&15; K halves 0-7/16-23 (lanes<16)
  // or 8-15/24-31 (lanes>=16) -> two contiguous 16B chunks.
  const unsigned short* xrow = xin + nlow * KD + lhalf * 8;
  const int NT = ND / 16, KT = KD / 32;
#pragma unroll 2
  for (int nt = 0; nt < NT; ++nt) {
    float bv = bias[nt * 16 + nlow];            // bias depends only on N
    v8f acc = {bv, bv, bv, bv, bv, bv, bv, bv};
    const unsigned short* wt = wp + nt * 512 + lane * 16;
    for (int kt = 0; kt < KT; ++kt) {
      BF16Frag Af, Bf;
      Af.h[0] = *(const us8*)(xrow + kt * 32);
      Af.h[1] = *(const us8*)(xrow + kt * 32 + 16);
      Bf.h[0] = *(const us8*)(wt + (size_t)kt * NT * 512);
      Bf.h[1] = *(const us8*)(wt + (size_t)kt * NT * 512 + 8);
      acc = __builtin_amdgcn_wmma_f32_16x16x32_bf16(false, Af.v, false, Bf.v,
                                                    (short)0, acc, false, false);
    }
    const int mb = lhalf * 8;                   // D: vgpr r -> M = r + 8*(lane>=16)
#pragma unroll
    for (int r = 0; r < 8; ++r) {
      yout[(mb + r) * ND + nt * 16 + nlow] = f2bf(celu01(acc[r]));
    }
  }
}

// ---------------------------------------------------------------------------
// Kernel 3: routed expert MLP. One wave32 per (type, 16-atom group).
// ---------------------------------------------------------------------------
__global__ __launch_bounds__(32)
void expert_mlp_kernel(const float* __restrict__ aev,
                       const int* __restrict__ lists,
                       const int* __restrict__ counts,
                       const unsigned short* __restrict__ wpk,
                       const float* __restrict__ b0,
                       const float* __restrict__ b1,
                       const float* __restrict__ b2,
                       const float* __restrict__ b3,
                       float* __restrict__ eatom,
                       int natoms) {
  const int t = blockIdx.y;
  const int base = blockIdx.x * 16;
  const int cnt = counts[t];
  if (base >= cnt) return;                      // uniform exit, EXEC stays full
  const int lane = threadIdx.x;

  __shared__ int sidx[16];
  __shared__ __align__(16) unsigned short x0[16 * AEVD];
  __shared__ __align__(16) unsigned short a1[16 * H0];
  __shared__ __align__(16) unsigned short a2[16 * H1];
  __shared__ __align__(16) unsigned short a3[16 * H2];

  if (lane < 16) {
    int p = base + lane;
    sidx[lane] = (p < cnt) ? lists[t * natoms + p] : -1;
  }
  __syncthreads();

  // Gather 16 aev rows (f32) -> bf16 row-major in LDS (float4 loads).
  for (int q = lane; q < 16 * (AEVD / 4); q += 32) {
    int r  = q / (AEVD / 4);
    int c4 = q - r * (AEVD / 4);
    int a  = sidx[r];
    float4 v = make_float4(0.f, 0.f, 0.f, 0.f);
    if (a >= 0) v = *(const float4*)(aev + (size_t)a * AEVD + c4 * 4);
    ushort4 o;
    o.x = f2bf(v.x); o.y = f2bf(v.y); o.z = f2bf(v.z); o.w = f2bf(v.w);
    *(ushort4*)(&x0[r * AEVD + c4 * 4]) = o;
  }
  __syncthreads();

  const unsigned short* wt = wpk + (size_t)t * PK_T;
  mlp_layer<AEVD, H0>(x0, a1, wt + PK_L0, b0 + t * H0, lane);
  __syncthreads();
  mlp_layer<H0, H1>(a1, a2, wt + PK_L1, b1 + t * H1, lane);
  __syncthreads();
  mlp_layer<H1, H2>(a2, a3, wt + PK_L2, b2 + t * H2, lane);
  __syncthreads();

  // Final linear layer: [16 x 96] @ [96 x 1] (N padded to 16 with zeros).
  const int nlow  = lane & 15;
  const int lhalf = (lane >> 4) & 1;
  float bv = (nlow == 0) ? b3[t] : 0.f;
  v8f acc = {bv, bv, bv, bv, bv, bv, bv, bv};
  const unsigned short* xrow = a3 + nlow * H2 + lhalf * 8;
  const unsigned short* w3   = wt + PK_L3 + lane * 16;
#pragma unroll
  for (int kt = 0; kt < 3; ++kt) {
    BF16Frag Af, Bf;
    Af.h[0] = *(const us8*)(xrow + kt * 32);
    Af.h[1] = *(const us8*)(xrow + kt * 32 + 16);
    Bf.h[0] = *(const us8*)(w3 + kt * 512);
    Bf.h[1] = *(const us8*)(w3 + kt * 512 + 8);
    acc = __builtin_amdgcn_wmma_f32_16x16x32_bf16(false, Af.v, false, Bf.v,
                                                  (short)0, acc, false, false);
  }
  if (nlow == 0) {                              // column N=0 lives in lanes 0,16
    const int mb = lhalf * 8;
#pragma unroll
    for (int r = 0; r < 8; ++r) {
      int a = sidx[mb + r];
      if (a >= 0) eatom[a] = acc[r];            // unique slot per atom
    }
  }
}

// ---------------------------------------------------------------------------
// Kernel 4: per-molecule reduction over 64 atoms (fixed order -> deterministic)
// ---------------------------------------------------------------------------
__global__ void reduce_kernel(const float* __restrict__ eatom,
                              float* __restrict__ out, int nmol, int A) {
  int b = blockIdx.x * blockDim.x + threadIdx.x;
  if (b >= nmol) return;
  float s = 0.f;
  for (int j = 0; j < A; ++j) s += eatom[(size_t)b * A + j];
  out[b] = s;
}

// Fallback if output buffer can only hold species as one float slot each.
__global__ void species_to_float_kernel(const long long* __restrict__ sp,
                                        float* __restrict__ out, int n) {
  int i = blockIdx.x * blockDim.x + threadIdx.x;
  if (i < n) out[i] = (float)sp[i];
}

// ---------------------------------------------------------------------------
extern "C" void kernel_launch(void* const* d_in, const int* in_sizes, int n_in,
                              void* d_out, int out_size, void* d_ws, size_t ws_size,
                              hipStream_t stream) {
  const long long* species = (const long long*)d_in[0];
  const float* aev = (const float*)d_in[1];
  const float* W0 = (const float*)d_in[2];
  const float* b0 = (const float*)d_in[3];
  const float* W1 = (const float*)d_in[4];
  const float* b1 = (const float*)d_in[5];
  const float* W2 = (const float*)d_in[6];
  const float* b2 = (const float*)d_in[7];
  const float* W3 = (const float*)d_in[8];
  const float* b3 = (const float*)d_in[9];

  const int natoms = in_sizes[0];   // B*A = 131072
  const int A = 64;
  const int nmol = natoms / A;

  // Workspace carve-up (all regions 256B aligned)
  char* ws = (char*)d_ws;
  size_t off = 0;
  int* counts = (int*)(ws + off); off += 256;
  int* lists  = (int*)(ws + off); off += (size_t)NTYPES * natoms * sizeof(int);
  off = (off + 255) & ~(size_t)255;
  float* eatom = (float*)(ws + off); off += (size_t)natoms * sizeof(float);
  off = (off + 255) & ~(size_t)255;
  unsigned short* wpk = (unsigned short*)(ws + off);

  init_counts_kernel<<<1, 32, 0, stream>>>(counts);

  const int pk_elems = NTYPES * PK_T;
  pack_weights_kernel<<<(pk_elems + 255) / 256, 256, 0, stream>>>(W0, W1, W2, W3, wpk);

  bucket_kernel<<<(natoms + 255) / 256, 256, 0, stream>>>(species, counts, lists, natoms);

  dim3 grid((natoms + 15) / 16, NTYPES);
  expert_mlp_kernel<<<grid, 32, 0, stream>>>(aev, lists, counts, wpk,
                                             b0, b1, b2, b3, eatom, natoms);

  // Outputs: (species, energies) concatenated flat in return order.
  float* outf = (float*)d_out;
  float* ewr;
  if ((size_t)out_size >= (size_t)2 * natoms + (size_t)nmol) {
    // Room for raw int64 species bytes (2 float slots each) + energies.
    hipMemcpyAsync(d_out, d_in[0], (size_t)natoms * sizeof(long long),
                   hipMemcpyDeviceToDevice, stream);
    ewr = outf + (size_t)2 * natoms;
  } else {
    species_to_float_kernel<<<(natoms + 255) / 256, 256, 0, stream>>>(species, outf, natoms);
    ewr = outf + natoms;
  }
  reduce_kernel<<<(nmol + 255) / 256, 256, 0, stream>>>(eatom, ewr, nmol, A);
}
